// IndigoTransformerLayer_5222680232487
// MI455X (gfx1250) — matrix operand order repaired and checked by reading, not compiled, not run on hardware
//
#include <hip/hip_runtime.h>

#define SDIM   1024
#define DMODEL 768
#define NHEAD  12
#define HDIM   64
#define DFF    3072
#define BATCH  2
#define NTOK   (BATCH*SDIM)   // 2048
#define D3     (3*DMODEL)     // 2304

typedef __attribute__((ext_vector_type(16))) __bf16 v16bf;
typedef __attribute__((ext_vector_type(8)))  float  v8f;

union FragAB { v16bf v; unsigned u[8]; };

// gfx1250 async copy (global -> LDS, ASYNCcnt) — guarded so compile never breaks
#if defined(__has_builtin)
#  if __has_builtin(__builtin_amdgcn_global_load_async_to_lds_b128) && \
      __has_builtin(__builtin_amdgcn_s_wait_asynccnt)
#    define HAS_ASYNC 1
#  endif
#endif
#ifndef HAS_ASYNC
#  define HAS_ASYNC 0
#endif

// builtin expects: (int4 __device__ *, int4 __shared__ *, imm offset, imm cpol)
typedef int v4i __attribute__((vector_size(16)));
#define AS_GLOBAL_V4I(p) ((__attribute__((address_space(1))) v4i*)(p))
#define AS_LDS_V4I(p)    ((__attribute__((address_space(3))) v4i*)(p))

__device__ __forceinline__ unsigned short f2bf(float f) {
  unsigned u = __float_as_uint(f);
  u += 0x7FFFu + ((u >> 16) & 1u);          // round-to-nearest-even
  return (unsigned short)(u >> 16);
}
__device__ __forceinline__ float bf2f(unsigned short h) {
  return __uint_as_float(((unsigned)h) << 16);
}

// ---------------------------------------------------------------- fp32 -> bf16
__global__ void cvt_f32_bf16(const float* __restrict__ in,
                             unsigned short* __restrict__ out, int n) {
  int i = blockIdx.x * blockDim.x + threadIdx.x;
  int stride = gridDim.x * blockDim.x;
  for (; i < n; i += stride) out[i] = f2bf(in[i]);
}

// --------------------------------------------------- LayerNorm + cast to bf16
// one block (256 threads) per row of D=768
__global__ void __launch_bounds__(256) ln_to_bf16(
    const float* __restrict__ x, const float* __restrict__ g,
    const float* __restrict__ b, unsigned short* __restrict__ out) {
  const int row = blockIdx.x, tid = threadIdx.x;
  const float* xr = x + (size_t)row * DMODEL;
  float v0 = xr[tid], v1 = xr[tid + 256], v2 = xr[tid + 512];
  float s  = v0 + v1 + v2;
  float s2 = v0 * v0 + v1 * v1 + v2 * v2;
  for (int m = 1; m < 32; m <<= 1) { s += __shfl_xor(s, m); s2 += __shfl_xor(s2, m); }
  __shared__ float ss[8], ssq[8];
  if ((tid & 31) == 0) { ss[tid >> 5] = s; ssq[tid >> 5] = s2; }
  __syncthreads();
  s = ss[tid & 7]; s2 = ssq[tid & 7];
  for (int m = 1; m < 8; m <<= 1) { s += __shfl_xor(s, m); s2 += __shfl_xor(s2, m); }
  float mean = s * (1.f / DMODEL);
  float var  = s2 * (1.f / DMODEL) - mean * mean;
  float rstd = rsqrtf(var + 1e-5f);
  unsigned short* outr = out + (size_t)row * DMODEL;
  outr[tid]       = f2bf((v0 - mean) * rstd * g[tid]       + b[tid]);
  outr[tid + 256] = f2bf((v1 - mean) * rstd * g[tid + 256] + b[tid + 256]);
  outr[tid + 512] = f2bf((v2 - mean) * rstd * g[tid + 512] + b[tid + 512]);
}

// ------------------------------------------------------- WMMA GEMM  C = A*B^T
// A: (M,K) bf16 row-major, Bw: (N,K) bf16 row-major.
// Block = 32-row band x 256 cols (8 waves x 32). A tile staged to LDS with
// async global->LDS copies (double buffered); B fragments streamed from global.
template<bool RELU, bool OUT_BF16>
__global__ void __launch_bounds__(256) gemm_bf16nt(
    const unsigned short* __restrict__ A, const unsigned short* __restrict__ Bw,
    const float* __restrict__ bias, const float* __restrict__ resid,
    void* __restrict__ outp, int M, int N, int K, int nblk_n) {
  const int lane = threadIdx.x & 31;
  const int wv   = threadIdx.x >> 5;
  const int m0 = (blockIdx.x / nblk_n) * 32;
  const int n0 = (blockIdx.x % nblk_n) * 256 + wv * 32;
  const int half = lane >> 4, lx = lane & 15;

  __shared__ unsigned short atile[2][32 * 32];   // 2 x 2KB

  // cooperative stage of A tile (32 rows x 32 k) for k-step k0 into buffer buf
  auto stageA = [&](int buf, int k0) {
    const int t = threadIdx.x;
    if (t < 128) {
      const int row = t >> 2, c16 = t & 3;                 // 4 x 16B per 64B row
      const unsigned short* gp = A + (size_t)(m0 + row) * K + k0 + c16 * 8;
      unsigned short* lp = &atile[buf][row * 32 + c16 * 8];
#if HAS_ASYNC
      __builtin_amdgcn_global_load_async_to_lds_b128(
          AS_GLOBAL_V4I(gp), AS_LDS_V4I(lp), 0, 0);
#else
      *reinterpret_cast<uint4*>(lp) = *reinterpret_cast<const uint4*>(gp);
#endif
    }
  };

  v8f c[2][2] = {};
  const unsigned short* brow[2] = { Bw + (size_t)(n0 + lx) * K,
                                    Bw + (size_t)(n0 + 16 + lx) * K };
  int kB[8];
#pragma unroll
  for (int v = 0; v < 8; ++v) kB[v] = (v < 4) ? (2 * v + 8 * half) : (16 + 2 * (v - 4) + 8 * half);

  stageA(0, 0);
  int buf = 0;
  for (int k0 = 0; k0 < K; k0 += 32) {
#if HAS_ASYNC
    __builtin_amdgcn_s_wait_asynccnt(0);
#endif
    __syncthreads();
    if (k0 + 32 < K) stageA(buf ^ 1, k0 + 32);

    FragAB a[2], bf[2];
    const unsigned short* at = atile[buf];
#pragma unroll
    for (int i = 0; i < 2; ++i) {
      const int row = i * 16 + lx;
      uint4 lo = *reinterpret_cast<const uint4*>(at + row * 32 + 8 * half);
      uint4 hi = *reinterpret_cast<const uint4*>(at + row * 32 + 16 + 8 * half);
      a[i].u[0] = lo.x; a[i].u[1] = lo.y; a[i].u[2] = lo.z; a[i].u[3] = lo.w;
      a[i].u[4] = hi.x; a[i].u[5] = hi.y; a[i].u[6] = hi.z; a[i].u[7] = hi.w;
    }
#pragma unroll
    for (int i = 0; i < 2; ++i)
#pragma unroll
      for (int v = 0; v < 8; ++v)
        bf[i].u[v] = *reinterpret_cast<const unsigned*>(brow[i] + k0 + kB[v]);
#pragma unroll
    for (int i = 0; i < 2; ++i)
#pragma unroll
      for (int j = 0; j < 2; ++j)
        c[i][j] = __builtin_amdgcn_wmma_f32_16x16x32_bf16(
            false, a[i].v, false, bf[j].v, (short)0, c[i][j], false, false);
    __syncthreads();
    buf ^= 1;
  }
#pragma unroll
  for (int i = 0; i < 2; ++i)
#pragma unroll
    for (int j = 0; j < 2; ++j)
#pragma unroll
      for (int v = 0; v < 8; ++v) {
        int gm = m0 + i * 16 + v + 8 * half;
        int gn = n0 + j * 16 + lx;
        float val = c[i][j][v];
        if (bias)  val += bias[gn];
        if (RELU)  val = fmaxf(val, 0.f);
        if (resid) val += resid[(size_t)gm * N + gn];
        if (OUT_BF16) ((unsigned short*)outp)[(size_t)gm * N + gn] = f2bf(val);
        else          ((float*)outp)[(size_t)gm * N + gn] = val;
      }
}

// -------------------------------------- V^T staging: vt[b, f, s] = V[b, s, f]
__global__ void transpose_v(const unsigned short* __restrict__ qkv,
                            unsigned short* __restrict__ vt) {
  int idx = blockIdx.x * blockDim.x + threadIdx.x;
  const int total = BATCH * DMODEL * SDIM;
  for (; idx < total; idx += gridDim.x * blockDim.x) {
    int s = idx & (SDIM - 1);
    int rest = idx >> 10;
    int f = rest % DMODEL;
    int b = rest / DMODEL;
    vt[idx] = qkv[(size_t)(b * SDIM + s) * D3 + 2 * DMODEL + f];
  }
}

// ---------------------------------------- flash attention with rel-pos gather
// one wave = 16 query rows of one (b,h); online softmax over 32-key blocks
__global__ void __launch_bounds__(256) attn_kernel(
    const unsigned short* __restrict__ qkv, const unsigned short* __restrict__ vt,
    const float* __restrict__ relA, const int* __restrict__ relM,
    const unsigned char* __restrict__ mask, unsigned short* __restrict__ out) {
  const int lane = threadIdx.x & 31;
  const int wv   = threadIdx.x >> 5;
  const int gid  = blockIdx.x * 8 + wv;
  const int qt = gid & 63;               // S/16 = 64 query tiles
  const int h  = (gid >> 6) % NHEAD;
  const int b  = gid / (64 * NHEAD);
  const int q0 = qt * 16;
  const int half = lane >> 4, lx = lane & 15;

  __shared__ float          rbuf[8][16][4];
  __shared__ unsigned short pbuf[8][16][32];

  // r[m][c] = q_row_m . rel_A[c]   (fp32, tiny)
  if (lane < 16) {
    const unsigned short* qrow = qkv + (size_t)(b * SDIM + q0 + lane) * D3 + h * HDIM;
    float r0 = 0.f, r1 = 0.f, r2 = 0.f;
    for (int d = 0; d < HDIM; ++d) {
      float qv = bf2f(qrow[d]);
      r0 += qv * relA[d];
      r1 += qv * relA[HDIM + d];
      r2 += qv * relA[2 * HDIM + d];
    }
    rbuf[wv][lane][0] = r0; rbuf[wv][lane][1] = r1; rbuf[wv][lane][2] = r2;
  }
  __syncthreads();

  FragAB aq[2];                                  // q fragment, features 0..31 / 32..63
  {
    const unsigned short* qrow = qkv + (size_t)(b * SDIM + q0 + lx) * D3 + h * HDIM;
#pragma unroll
    for (int v = 0; v < 8; ++v) {
      int kA = 2 * v + 16 * half;
      aq[0].u[v] = *reinterpret_cast<const unsigned*>(qrow + kA);
      aq[1].u[v] = *reinterpret_cast<const unsigned*>(qrow + 32 + kA);
    }
  }

  v8f acc[4] = {};
  float mrow[8], lrow[8];
#pragma unroll
  for (int v = 0; v < 8; ++v) { mrow[v] = -1e30f; lrow[v] = 0.f; }

  const int* relB = relM + (size_t)b * SDIM * SDIM;

  for (int kt = 0; kt < SDIM; kt += 32) {
    v8f c0 = {}, c1 = {};
    {
      const unsigned short* krow0 = qkv + (size_t)(b * SDIM + kt + lx) * D3 + DMODEL + h * HDIM;
      const unsigned short* krow1 = krow0 + (size_t)16 * D3;
      FragAB bk0a, bk0b, bk1a, bk1b;
#pragma unroll
      for (int v = 0; v < 8; ++v) {
        int kB = 2 * v + 16 * half;
        bk0a.u[v] = *reinterpret_cast<const unsigned*>(krow0 + kB);
        bk0b.u[v] = *reinterpret_cast<const unsigned*>(krow0 + 32 + kB);
        bk1a.u[v] = *reinterpret_cast<const unsigned*>(krow1 + kB);
        bk1b.u[v] = *reinterpret_cast<const unsigned*>(krow1 + 32 + kB);
      }
      c0 = __builtin_amdgcn_wmma_f32_16x16x32_bf16(false, aq[0].v, false, bk0a.v, (short)0, c0, false, false);
      c0 = __builtin_amdgcn_wmma_f32_16x16x32_bf16(false, aq[1].v, false, bk0b.v, (short)0, c0, false, false);
      c1 = __builtin_amdgcn_wmma_f32_16x16x32_bf16(false, aq[0].v, false, bk1a.v, (short)0, c1, false, false);
      c1 = __builtin_amdgcn_wmma_f32_16x16x32_bf16(false, aq[1].v, false, bk1b.v, (short)0, c1, false, false);
    }
    const int key0 = kt + lx, key1 = kt + 16 + lx;
    const float msk0 = mask[b * SDIM + key0] ? -1e30f : 0.f;
    const float msk1 = mask[b * SDIM + key1] ? -1e30f : 0.f;
    float alpha[8];
#pragma unroll
    for (int v = 0; v < 8; ++v) {
      const int m  = v + 8 * half;
      const int tq = q0 + m;
      const int* rr = relB + (size_t)tq * SDIM;
      int i0 = rr[key0] + 1;                       // {-1,0,1} -> {0,1,2}
      int i1 = rr[key1] + 1;
      float l0 = (c0[v] + rbuf[wv][m][i0]) * 0.125f + msk0;
      float l1 = (c1[v] + rbuf[wv][m][i1]) * 0.125f + msk1;
      float t = fmaxf(l0, l1);
      for (int ms = 1; ms < 16; ms <<= 1) t = fmaxf(t, __shfl_xor(t, ms));
      float mn = fmaxf(mrow[v], t);
      float al = __expf(mrow[v] - mn);
      float e0 = __expf(l0 - mn);
      float e1 = __expf(l1 - mn);
      float rs = e0 + e1;
      for (int ms = 1; ms < 16; ms <<= 1) rs += __shfl_xor(rs, ms);
      lrow[v] = lrow[v] * al + rs;
      mrow[v] = mn;
      alpha[v] = al;
      pbuf[wv][m][lx]      = f2bf(e0);
      pbuf[wv][m][16 + lx] = f2bf(e1);
    }
#pragma unroll
    for (int f = 0; f < 4; ++f)
#pragma unroll
      for (int v = 0; v < 8; ++v) acc[f][v] *= alpha[v];
    __syncthreads();
    FragAB ap;                                     // P re-striped to A-layout
#pragma unroll
    for (int v = 0; v < 8; ++v) {
      int kA = 2 * v + 16 * half;
      ap.u[v] = *reinterpret_cast<const unsigned*>(&pbuf[wv][lx][kA]);
    }
#pragma unroll
    for (int f = 0; f < 4; ++f) {
      FragAB bv;
      const unsigned short* vrow =
          vt + (size_t)(b * DMODEL + h * HDIM + f * 16 + lx) * SDIM + kt;
#pragma unroll
      for (int v = 0; v < 8; ++v) {
        int kB = 2 * v + 16 * half;
        bv.u[v] = *reinterpret_cast<const unsigned*>(vrow + kB);
      }
      acc[f] = __builtin_amdgcn_wmma_f32_16x16x32_bf16(
          false, ap.v, false, bv.v, (short)0, acc[f], false, false);
    }
    __syncthreads();
  }
#pragma unroll
  for (int f = 0; f < 4; ++f)
#pragma unroll
    for (int v = 0; v < 8; ++v) {
      int m = v + 8 * half;
      float o = acc[f][v] / lrow[v];
      out[(size_t)(b * SDIM + q0 + m) * DMODEL + h * HDIM + f * 16 + lx] = f2bf(o);
    }
}

// ------------------------------------------------------------------- launcher
extern "C" void kernel_launch(void* const* d_in, const int* in_sizes, int n_in,
                              void* d_out, int out_size, void* d_ws, size_t ws_size,
                              hipStream_t stream) {
  (void)in_sizes; (void)n_in; (void)out_size; (void)ws_size;
  const float*         inp  = (const float*)d_in[0];
  const unsigned char* mask = (const unsigned char*)d_in[1];
  const int*           relM = (const int*)d_in[2];
  const float*         qkvw = (const float*)d_in[3];
  const float*         relA = (const float*)d_in[4];
  const float*         ow   = (const float*)d_in[5];
  const float*         w1   = (const float*)d_in[6];
  const float*         b1   = (const float*)d_in[7];
  const float*         w2   = (const float*)d_in[8];
  const float*         b2   = (const float*)d_in[9];
  const float*         ln1g = (const float*)d_in[10];
  const float*         ln1b = (const float*)d_in[11];
  const float*         ln2g = (const float*)d_in[12];
  const float*         ln2b = (const float*)d_in[13];
  float* out = (float*)d_out;

  char* ws = (char*)d_ws;
  auto alloc = [&](size_t bytes) -> void* {
    void* p = ws; ws += (bytes + 255) & ~(size_t)255; return p;
  };
  unsigned short* qkvw_bf = (unsigned short*)alloc((size_t)D3 * DMODEL * 2);
  unsigned short* ow_bf   = (unsigned short*)alloc((size_t)DMODEL * DMODEL * 2);
  unsigned short* w1_bf   = (unsigned short*)alloc((size_t)DFF * DMODEL * 2);
  unsigned short* w2_bf   = (unsigned short*)alloc((size_t)DMODEL * DFF * 2);
  unsigned short* xln     = (unsigned short*)alloc((size_t)NTOK * DMODEL * 2);
  unsigned short* qkv     = (unsigned short*)alloc((size_t)NTOK * D3 * 2);
  unsigned short* vt      = (unsigned short*)alloc((size_t)BATCH * DMODEL * SDIM * 2);
  unsigned short* aout    = (unsigned short*)alloc((size_t)NTOK * DMODEL * 2);
  float*          x1      = (float*)alloc((size_t)NTOK * DMODEL * 4);
  unsigned short* hln     = (unsigned short*)alloc((size_t)NTOK * DMODEL * 2);
  unsigned short* mlp1    = (unsigned short*)alloc((size_t)NTOK * DFF * 2);

  auto cvt = [&](const float* src, unsigned short* dst, int n) {
    cvt_f32_bf16<<<(n + 2047) / 2048, 256, 0, stream>>>(src, dst, n);
  };
  cvt(qkvw, qkvw_bf, D3 * DMODEL);
  cvt(ow,   ow_bf,   DMODEL * DMODEL);
  cvt(w1,   w1_bf,   DFF * DMODEL);
  cvt(w2,   w2_bf,   DMODEL * DFF);

  // LN1
  ln_to_bf16<<<NTOK, 256, 0, stream>>>(inp, ln1g, ln1b, xln);
  // QKV = ln1(x) @ qkv_w^T  (2048 x 2304 x 768)
  gemm_bf16nt<false, true><<<(NTOK / 32) * (D3 / 256), 256, 0, stream>>>(
      xln, qkvw_bf, nullptr, nullptr, qkv, NTOK, D3, DMODEL, D3 / 256);
  // stage V^T
  transpose_v<<<(BATCH * DMODEL * SDIM) / 256, 256, 0, stream>>>(qkv, vt);
  // attention: 2*12*64 = 1536 waves -> 192 blocks
  attn_kernel<<<BATCH * NHEAD * (SDIM / 16) / 8, 256, 0, stream>>>(
      qkv, vt, relA, relM, mask, aout);
  // x1 = inp + attn_out @ o_w^T
  gemm_bf16nt<false, false><<<(NTOK / 32) * (DMODEL / 256), 256, 0, stream>>>(
      aout, ow_bf, nullptr, inp, x1, NTOK, DMODEL, DMODEL, DMODEL / 256);
  // LN2
  ln_to_bf16<<<NTOK, 256, 0, stream>>>(x1, ln2g, ln2b, hln);
  // mlp1 = relu(h @ w1^T + b1)
  gemm_bf16nt<true, true><<<(NTOK / 32) * (DFF / 256), 256, 0, stream>>>(
      hln, w1_bf, b1, nullptr, mlp1, NTOK, DFF, DMODEL, DFF / 256);
  // out = x1 + mlp1 @ w2^T + b2
  gemm_bf16nt<false, false><<<(NTOK / 32) * (DMODEL / 256), 256, 0, stream>>>(
      mlp1, w2_bf, b2, x1, out, NTOK, DMODEL, DFF, DMODEL / 256);
}
